// HierarchicalTripletLoss_7928509628770
// MI455X (gfx1250) — matrix operand order repaired and checked by reading, not compiled, not run on hardware
//
#include <hip/hip_runtime.h>

typedef __attribute__((ext_vector_type(16))) __bf16 v16bf;
typedef __attribute__((ext_vector_type(8)))  __bf16 v8bf;
typedef __attribute__((ext_vector_type(4)))  __bf16 v4bf;
typedef __attribute__((ext_vector_type(8)))  float  v8f;
typedef __attribute__((ext_vector_type(4)))  unsigned int v4u;
typedef __attribute__((ext_vector_type(8)))  int    v8i;
typedef __attribute__((ext_vector_type(4)))  int    v4i;

#define NL     64
#define DF     1024
#define FSTR   1032      // padded bf16 LDS row stride (4-bank stride between rows)
#define DSTR   66        // padded D row stride (floats)
#define MARG   1.0f

// LDS layout (bytes):
#define OFF_F     0                              // 64*1032*2 = 132096
#define OFF_D     (NL * FSTR * 2)                // 64*66*4   = 16896
#define OFF_SCALE (OFF_D + NL * DSTR * 4)        // 256
#define OFF_RED   (OFF_SCALE + NL * 4)           // 2048
#define OFF_LAB   (OFF_RED + 512 * 4)            // 256
#define OFF_STAGE (OFF_LAB + NL * 4)             // 32*1024*4 = 131072
#define SMEM_TOTAL (OFF_STAGE + 32 * DF * 4)     // 282624 B

__device__ __forceinline__ int level_of(int l) { return l < 4 ? 0 : (l < 16 ? 1 : 2); }

// Issue one TDM load: 32 rows x 1024 f32 from `gaddr` -> LDS byte offset `ldsOff`.
__device__ __forceinline__ void tdm_load_half(const float* gaddr, unsigned ldsOff)
{
    const unsigned long long ga = (unsigned long long)(uintptr_t)gaddr;
    // D# group 0: count=1 | lds_addr | global_addr[56:0] | type=2 (bits 127:126)
    v4u g0 = { 1u, ldsOff, (unsigned)ga,
               ((unsigned)(ga >> 32) & 0x01FFFFFFu) | 0x80000000u };
    // D# group 1: data_size=2 (4B); tensor_dim0=1024; tensor_dim1=64;
    //             tile_dim0=1024; tile_dim1=32; tensor_dim0_stride=1024
    v8i g1 = { (int)(2u << 16),          // mask=0, data_size=4B, no flags
               (int)(1024u << 16),       // barrier_addr=0 | tensor_dim0.lo16
               (int)(64u << 16),         // tensor_dim0.hi16=0 | tensor_dim1.lo16
               (int)(1024u << 16),       // tensor_dim1.hi16=0 | tile_dim0
               (int)32u,                 // tile_dim1=32 | tile_dim2=0
               (int)1024u,               // tensor_dim0_stride[31:0]
               0, 0 };                   // stride hi / dim1_stride
    v4i gz = { 0, 0, 0, 0 };
#if __clang_major__ >= 23
    v8i gz8 = { 0, 0, 0, 0, 0, 0, 0, 0 };
    __builtin_amdgcn_tensor_load_to_lds(g0, g1, gz, gz, gz8, 0);
#else
    __builtin_amdgcn_tensor_load_to_lds(g0, g1, gz, gz, 0);
#endif
}

extern "C" __global__ __launch_bounds__(256)
void httl_main(const float* __restrict__ feats, const int* __restrict__ labels,
               float* __restrict__ partial)
{
    extern __shared__ char smem_raw[];
    __bf16* sF     = (__bf16*)(smem_raw + OFF_F);
    float*  sD     = (float*)(smem_raw + OFF_D);
    float*  sScale = (float*)(smem_raw + OFF_SCALE);
    float*  sRed   = (float*)(smem_raw + OFF_RED);
    int*    sLab   = (int*)(smem_raw + OFF_LAB);
    float*  sStage = (float*)(smem_raw + OFF_STAGE);

    const int b    = blockIdx.x;
    const int tid  = threadIdx.x;
    const int lane = tid & 31;
    const int wv   = tid >> 5;
    const float* f = feats + (size_t)b * NL * DF;

    if (tid < NL) sLab[tid] = labels[b * NL + tid];

    // ---- Phase 1: TDM-stage each half-batch, then normalize -> bf16 in LDS ----
    for (int h = 0; h < 2; ++h) {
        if (wv == 0) {                                   // one TDM op per workgroup
            tdm_load_half(f + (size_t)h * 32 * DF, (unsigned)OFF_STAGE);
            __builtin_amdgcn_s_wait_tensorcnt(0);        // TENSORcnt is per-wave
        }
        __syncthreads();                                 // publish staging buffer

        // per-row L2 norms from LDS (4 staged rows per wave)
        for (int r = 0; r < 4; ++r) {
            const int srow = wv * 4 + r;
            const float4* fr = (const float4*)(sStage + srow * DF);
            float s = 0.f;
            for (int i = lane; i < DF / 4; i += 32) {
                float4 v = fr[i];
                s += v.x * v.x + v.y * v.y + v.z * v.z + v.w * v.w;
            }
            for (int off = 16; off; off >>= 1) s += __shfl_xor(s, off, 32);
            if (lane == 0) sScale[h * 32 + srow] = 1.0f / fmaxf(sqrtf(s), 1e-12f);
        }
        __syncthreads();

        // scale + convert to bf16, store into padded WMMA layout
        for (int i = tid; i < (32 * DF) / 4; i += 256) {
            const int srow = i >> 8;                     // 256 float4 per row
            const int col  = (i & 255) * 4;
            float4 v = ((const float4*)sStage)[i];
            const float sc = sScale[h * 32 + srow];
            v4bf p;
            p[0] = (__bf16)(v.x * sc);
            p[1] = (__bf16)(v.y * sc);
            p[2] = (__bf16)(v.z * sc);
            p[3] = (__bf16)(v.w * sc);
            *(v4bf*)(sF + (h * 32 + srow) * FSTR + col) = p;
        }
        __syncthreads();                                 // staging reusable next half
    }

    // ---- Phase 2: Gram via v_wmma_f32_16x16x32_bf16; B shared by both tiles ----
    const int half = lane >> 4;   // lane group (ISA 7.12.2 16-bit layouts)
    const int mm   = lane & 15;
    {
        const int tn  = wv & 3;
        const int tm0 = wv >> 2;          // 0 or 1
        const int tm1 = tm0 + 2;          // 2 or 3
        const __bf16* a0row = sF + (tm0 * 16 + mm) * FSTR;
        const __bf16* a1row = sF + (tm1 * 16 + mm) * FSTR;
        const __bf16* brow  = sF + (tn  * 16 + mm) * FSTR;
        v8f acc0 = {}, acc1 = {};
        for (int k0 = 0; k0 < DF; k0 += 32) {
            // B 32x16 bf16: elems 0..7 -> K=k0+8*half ; elems 8..15 -> K=k0+16+8*half (shared)
            v8bf b0 = *(const v8bf*)(brow + k0 + half * 8);
            v8bf b1 = *(const v8bf*)(brow + k0 + 16 + half * 8);
            // A 16x32 bf16: elems 0..7 -> K=k0+8*half ; elems 8..15 -> K=k0+16+8*half
            v8bf p0 = *(const v8bf*)(a0row + k0 + half * 8);
            v8bf p1 = *(const v8bf*)(a0row + k0 + 16 + half * 8);
            v8bf q0 = *(const v8bf*)(a1row + k0 + half * 8);
            v8bf q1 = *(const v8bf*)(a1row + k0 + 16 + half * 8);
            v16bf B, A0, A1;
            for (int e = 0; e < 8; ++e) {
                B[e] = b0[e];  B[e + 8] = b1[e];
                A0[e] = p0[e]; A0[e + 8] = p1[e];
                A1[e] = q0[e]; A1[e + 8] = q1[e];
            }
            acc0 = __builtin_amdgcn_wmma_f32_16x16x32_bf16(
                       false, A0, false, B, (short)0, acc0, false, false);
            acc1 = __builtin_amdgcn_wmma_f32_16x16x32_bf16(
                       false, A1, false, B, (short)0, acc1, false, false);
        }
        // C/D layout: VGPR r -> M = r + 8*half ; N = lane&15
        for (int r = 0; r < 8; ++r) {
            sD[(tm0 * 16 + r + half * 8) * DSTR + tn * 16 + mm] = 1.0f - acc0[r];
            sD[(tm1 * 16 + r + half * 8) * DSTR + tn * 16 + mm] = 1.0f - acc1[r];
        }
    }
    __syncthreads();

    // ---- Phase 3: sparse masked triplet reduction ----
    float s = 0.f, c = 0.f;
    for (int idx = tid; idx < NL * NL; idx += 256) {
        const int a = idx >> 6, n = idx & 63;
        if (a < 4) continue;                       // roots have no ancestors
        if (!sLab[a] || !sLab[n]) continue;
        int p0, p1;
        if (a < 16) { p0 = (a - 4) / 3; p1 = -1; } // level-1: parent root
        else        { p0 = 4 + (a - 16) / 4; p1 = (p0 - 4) / 3; } // leaf: child, root
        if (n == p0 || n == p1) continue;          // n must not be an ancestor of a
        const float ld   = fabsf((float)(level_of(a) - level_of(n)));
        const float base = MARG * ld - sD[a * DSTR + n];
        if (sLab[p0])            { s += fmaxf(sD[a * DSTR + p0] + base, 0.f); c += 1.f; }
        if (p1 >= 0 && sLab[p1]) { s += fmaxf(sD[a * DSTR + p1] + base, 0.f); c += 1.f; }
    }
    sRed[tid] = s; sRed[256 + tid] = c;
    __syncthreads();
    for (int off = 128; off; off >>= 1) {
        if (tid < off) { sRed[tid] += sRed[tid + off];
                         sRed[256 + tid] += sRed[256 + tid + off]; }
        __syncthreads();
    }
    if (tid == 0) { partial[2 * b] = sRed[0]; partial[2 * b + 1] = sRed[256]; }
}

extern "C" __global__ __launch_bounds__(256)
void httl_final(const float* __restrict__ partial, float* __restrict__ out)
{
    __shared__ float rs[256], rc[256];
    const int tid = threadIdx.x;
    rs[tid] = partial[2 * tid];
    rc[tid] = partial[2 * tid + 1];
    __syncthreads();
    for (int off = 128; off; off >>= 1) {
        if (tid < off) { rs[tid] += rs[tid + off]; rc[tid] += rc[tid + off]; }
        __syncthreads();
    }
    if (tid == 0) {
        const float loss = rs[0], cnt = rc[0];
        out[0] = (cnt > 0.f) ? loss / fmaxf(cnt, 1.f) : loss;
    }
}

extern "C" void kernel_launch(void* const* d_in, const int* in_sizes, int n_in,
                              void* d_out, int out_size, void* d_ws, size_t ws_size,
                              hipStream_t stream) {
    (void)in_sizes; (void)n_in; (void)out_size; (void)ws_size;
    const float* feats  = (const float*)d_in[0];
    const int*   labels = (const int*)d_in[1];
    float*       part   = (float*)d_ws;                 // 256 * {loss, count}

    (void)hipFuncSetAttribute((const void*)httl_main,
                              hipFuncAttributeMaxDynamicSharedMemorySize,
                              (int)SMEM_TOTAL);
    httl_main<<<256, 256, SMEM_TOTAL, stream>>>(feats, labels, part);
    httl_final<<<1, 256, 0, stream>>>(part, (float*)d_out);
}
